// Apriel2KDAMixer_45621142618778
// MI455X (gfx1250) — matrix operand order repaired and checked by reading, not compiled, not run on hardware
//
#include <hip/hip_runtime.h>
#include <hip/hip_bf16.h>

#define T_   4096
#define HID_ 4096
#define H_   32
#define D_   64
#define P_   2048

typedef __attribute__((ext_vector_type(16))) __bf16 v16bf;
typedef __attribute__((ext_vector_type(8)))  float  v8f;

// --------------------------------------------------------- async-LDS helpers
// GLOBAL_LOAD_ASYNC_TO_LDS_B128 (GV mode): per-lane copy of 16B global -> LDS,
// tracked by ASYNCcnt (CDNA5 ISA).
__device__ __forceinline__ void async_copy_b128(unsigned lds_off, const void* gptr) {
    unsigned long long ga = (unsigned long long)(uintptr_t)gptr;
    asm volatile("global_load_async_to_lds_b128 %0, %1, off"
                 :: "v"(lds_off), "v"(ga) : "memory");
}
__device__ __forceinline__ void wait_async0() {
    asm volatile("s_wait_asynccnt 0x0" ::: "memory");
}

// ---------------------------------------------------------------- elementwise
__global__ void k_cvt_bf16(const float* __restrict__ s,
                           __hip_bfloat16* __restrict__ d, int n) {
    int i = blockIdx.x * blockDim.x + threadIdx.x;
    if (i < n) d[i] = __float2bfloat16(s[i]);
}

// causal depthwise conv (K=4) + SiLU, over [T,P]
__global__ void k_conv_silu(const float* __restrict__ x,
                            const float* __restrict__ w,
                            float* __restrict__ y) {
    int i = blockIdx.x * blockDim.x + threadIdx.x;   // T_*P_ threads
    int p = i & (P_ - 1);
    int t = i >> 11;
    float s = 0.f;
#pragma unroll
    for (int j = 0; j < 4; j++) {
        int tt = t - 3 + j;
        if (tt >= 0) s += x[(size_t)tt * P_ + p] * w[p * 4 + j];
    }
    y[i] = s / (1.f + __expf(-s));                   // silu
}

// g = -exp(A_log[h]) * softplus(g_lin + dt_bias)
__global__ void k_gate(float* __restrict__ g,
                       const float* __restrict__ dt_bias,
                       const float* __restrict__ A_log) {
    int i = blockIdx.x * blockDim.x + threadIdx.x;   // T_*P_
    int p = i & (P_ - 1);
    int h = p >> 6;
    float v = g[i] + dt_bias[p];
    float sp = (v > 20.f) ? v : log1pf(__expf(v));
    g[i] = -__expf(A_log[h]) * sp;
}

// L2-normalize each 64-vector row of [T*H, 64]; one wave per row
__global__ __launch_bounds__(256)
void k_l2norm(float* __restrict__ v, float scale) {
    int row  = blockIdx.x * 8 + (threadIdx.x >> 5);
    int lane = threadIdx.x & 31;
    size_t base = (size_t)row * 64;
    float a = v[base + lane], b = v[base + lane + 32];
    float s = a * a + b * b;
#pragma unroll
    for (int m = 16; m >= 1; m >>= 1) s += __shfl_xor(s, m, 32);
    float r = rsqrtf(s + 1e-6f) * scale;
    v[base + lane]      = a * r;
    v[base + lane + 32] = b * r;
}

// gated RMSNorm -> bf16; one wave per (t,h) row
__global__ __launch_bounds__(256)
void k_outnorm(const float* __restrict__ o, const float* __restrict__ g2,
               const float* __restrict__ w, __hip_bfloat16* __restrict__ ob) {
    int row  = blockIdx.x * 8 + (threadIdx.x >> 5);
    int lane = threadIdx.x & 31;
    size_t base = (size_t)row * 64;
    float a = o[base + lane], b = o[base + lane + 32];
    float s = a * a + b * b;
#pragma unroll
    for (int m = 16; m >= 1; m >>= 1) s += __shfl_xor(s, m, 32);
    float rms = rsqrtf(s * (1.f / 64.f) + 1e-5f);
    float ga = 1.f / (1.f + __expf(-g2[base + lane]));
    float gb = 1.f / (1.f + __expf(-g2[base + lane + 32]));
    ob[base + lane]      = __float2bfloat16(a * rms * w[lane] * ga);
    ob[base + lane + 32] = __float2bfloat16(b * rms * w[lane + 32] * gb);
}

// ---------------------------------------------------------------- WMMA GEMM
// C[M,N] (f32) = A[M,K] x B[K,N], A/B bf16 row-major.
// 256 threads / 8 waves; block tile 128x128x32; wave tile 32x64 (2x4 wmma).
// Double-buffered LDS; A staged with global_load_async_to_lds_b128 (ASYNCcnt),
// B staged through VGPRs with an LDS-transpose so fragment pairs are b32 loads.
__global__ __launch_bounds__(256)
void k_gemm_bf16(const __hip_bfloat16* __restrict__ A,
                 const __hip_bfloat16* __restrict__ B,
                 float* __restrict__ C, int M, int N, int Kd) {
    __shared__ unsigned short Al[2][128][32];   // [buf][m][k]
    __shared__ unsigned short Bt[2][128][32];   // [buf][n][k] (transposed)

    const int tid  = threadIdx.x;
    const int lane = tid & 31;
    const int wid  = tid >> 5;
    const int wm   = wid & 3;                // 4 waves along M
    const int wn   = wid >> 2;               // 2 waves along N
    const int m0   = blockIdx.y * 128;
    const int n0   = blockIdx.x * 128;

    v8f acc[2][4];
#pragma unroll
    for (int i = 0; i < 2; i++)
#pragma unroll
        for (int j = 0; j < 4; j++) acc[i][j] = (v8f){0.f,0.f,0.f,0.f,0.f,0.f,0.f,0.f};

    const int arow = tid >> 1, aseg = tid & 1;   // A: 128 rows x 2 halves of 16 elems
    const int brow = tid >> 3, bseg = tid & 7;   // B: 32 rows  x 8 segs   of 16 elems
    const int half = lane >> 4, l16 = lane & 15;

    const unsigned abase[2] = { (unsigned)(uintptr_t)&Al[0][arow][aseg * 16],
                                (unsigned)(uintptr_t)&Al[1][arow][aseg * 16] };
    const __hip_bfloat16* agp = A + (size_t)(m0 + arow) * Kd + aseg * 16;
    const __hip_bfloat16* bgp = B + (size_t)brow * N + n0 + bseg * 16;

    // ---- prologue: stage tile 0
    async_copy_b128(abase[0],      agp);
    async_copy_b128(abase[0] + 16, agp + 8);
    {
        const uint4* src = (const uint4*)bgp;
        union { uint4 q[2]; unsigned short s[16]; } u;
        u.q[0] = src[0]; u.q[1] = src[1];
#pragma unroll
        for (int e = 0; e < 16; e++) Bt[0][bseg * 16 + e][brow] = u.s[e];
    }
    wait_async0();
    __syncthreads();

    int cur = 0;
    for (int k0 = 0; k0 < Kd; k0 += 32) {
        const int nxt = cur ^ 1;
        const bool more = (k0 + 32) < Kd;
        uint4 bq0, bq1;
        if (more) {
            // issue async A copy for next tile (no VGPR round-trip)
            async_copy_b128(abase[nxt],      agp + (k0 + 32));
            async_copy_b128(abase[nxt] + 16, agp + (k0 + 32) + 8);
            // start B global loads for next tile into registers
            const uint4* src = (const uint4*)(bgp + (size_t)(k0 + 32) * N);
            bq0 = src[0]; bq1 = src[1];
            __builtin_prefetch(agp + (k0 + 64), 0, 1);
        }

        // ---- compute current tile
        // B fragments: VGPR v holds K=2v,2v+1 (lanes 0-15) / K=16+2v,2v+1 (16-31)
        v16bf bfrag[4];
#pragma unroll
        for (int tj = 0; tj < 4; tj++) {
            union { unsigned int u[8]; v16bf v; } ub;
            int n = wn * 64 + tj * 16 + l16;
#pragma unroll
            for (int p = 0; p < 8; p++)
                ub.u[p] = *(const unsigned int*)&Bt[cur][n][2 * p + half * 16];
            bfrag[tj] = ub.v;
        }
        // A fragments: VGPR p -> K=2p,2p+1 (lanes 0-15) / K=16+2p,2p+1 (16-31)
#pragma unroll
        for (int ti = 0; ti < 2; ti++) {
            union { unsigned int u[8]; v16bf v; } ua;
            int r = wm * 32 + ti * 16 + l16;
#pragma unroll
            for (int p = 0; p < 8; p++) {
                int kb = 2 * p + half * 16;
                ua.u[p] = *(const unsigned int*)&Al[cur][r][kb];
            }
#pragma unroll
            for (int tj = 0; tj < 4; tj++)
                acc[ti][tj] = __builtin_amdgcn_wmma_f32_16x16x32_bf16(
                    false, ua.v, false, bfrag[tj], (short)0, acc[ti][tj], false, false);
        }

        if (more) {
            // scatter-transpose B registers into next buffer
            union { uint4 q[2]; unsigned short s[16]; } u;
            u.q[0] = bq0; u.q[1] = bq1;
#pragma unroll
            for (int e = 0; e < 16; e++) Bt[nxt][bseg * 16 + e][brow] = u.s[e];
            wait_async0();      // this wave's async A copies into nxt are done
            __syncthreads();    // all waves done writing nxt / reading cur
        }
        cur = nxt;
    }

    // C/D layout: VGPR r -> M=r (lanes 0-15) / M=8+r (lanes 16-31), N = lane%16
#pragma unroll
    for (int ti = 0; ti < 2; ti++)
#pragma unroll
        for (int tj = 0; tj < 4; tj++) {
            int col  = n0 + wn * 64 + tj * 16 + l16;
            int rowb = m0 + wm * 32 + ti * 16 + half * 8;
#pragma unroll
            for (int r = 0; r < 8; r++)
                C[(size_t)(rowb + r) * N + col] = acc[ti][tj][r];
        }
}

// ---------------------------------------------------------------- small GEMMs
// C[M,N] = A[M,K] x B[K,N], N<=64 (32 or 64), f32. act=1 -> sigmoid.
__global__ __launch_bounds__(256)
void k_skinny(const float* __restrict__ A, const float* __restrict__ B,
              float* __restrict__ C, int Kd, int N, int act) {
    __shared__ float Al[64][65];
    __shared__ float Bl[64][64];
    int tid = threadIdx.x;
    int m0  = blockIdx.x * 64;
    float acc[16];
    int per = (64 * N) >> 8;
#pragma unroll
    for (int i = 0; i < 16; i++) acc[i] = 0.f;

    for (int k0 = 0; k0 < Kd; k0 += 64) {
#pragma unroll
        for (int i = 0; i < 16; i++) {
            int idx = tid * 16 + i;
            int r = idx >> 6, c = idx & 63;
            Al[r][c] = A[(size_t)(m0 + r) * Kd + k0 + c];
        }
        for (int idx = tid; idx < 64 * N; idx += 256) {
            int r = idx / N, c = idx % N;
            Bl[r][c] = B[(size_t)(k0 + r) * N + c];
        }
        __syncthreads();
        for (int i = 0; i < per; i++) {
            int idx = tid + (i << 8);
            int r = idx / N, c = idx % N;
            float s = acc[i];
#pragma unroll
            for (int kk = 0; kk < 64; kk++) s += Al[r][kk] * Bl[kk][c];
            acc[i] = s;
        }
        __syncthreads();
    }
    for (int i = 0; i < per; i++) {
        int idx = tid + (i << 8);
        int r = idx / N, c = idx % N;
        float v = acc[i];
        if (act == 1) v = 1.f / (1.f + __expf(-v));
        C[(size_t)(m0 + r) * N + c] = v;
    }
}

// C[T,P] = A[T,64] x B[64,P]; tile 16 rows x 128 cols per block
__global__ __launch_bounds__(256)
void k_expand(const float* __restrict__ A, const float* __restrict__ B,
              float* __restrict__ C) {
    __shared__ float Al[16][64];
    __shared__ float Bl[64][128];
    int tid = threadIdx.x;
    int t0 = blockIdx.y * 16;
    int p0 = blockIdx.x * 128;
#pragma unroll
    for (int i = 0; i < 4; i++) {
        int idx = tid + (i << 8);
        Al[idx >> 6][idx & 63] = A[(size_t)(t0 + (idx >> 6)) * 64 + (idx & 63)];
    }
#pragma unroll
    for (int i = 0; i < 32; i++) {
        int idx = tid + (i << 8);
        int r = idx >> 7, c = idx & 127;
        Bl[r][c] = B[(size_t)r * P_ + p0 + c];
    }
    __syncthreads();
    int c = tid & 127;
    int rbase = (tid >> 7) * 8;
    for (int rr = 0; rr < 8; rr++) {
        int r = rbase + rr;
        float s = 0.f;
#pragma unroll
        for (int kk = 0; kk < 64; kk++) s += Al[r][kk] * Bl[kk][c];
        C[(size_t)(t0 + r) * P_ + p0 + c] = s;
    }
}

// ---------------------------------------------------------------- KDA scan
// One wave per (head, 8 dv-columns): S[:,dv] chains are independent.
__global__ __launch_bounds__(32)
void k_scan(const float* __restrict__ q, const float* __restrict__ k,
            const float* __restrict__ v, const float* __restrict__ g,
            const float* __restrict__ beta, float* __restrict__ o) {
    int h   = blockIdx.x >> 3;
    int dv0 = (blockIdx.x & 7) * 8;
    int lane = threadIdx.x;

    float S[8][2];
#pragma unroll
    for (int c = 0; c < 8; c++) { S[c][0] = 0.f; S[c][1] = 0.f; }

    for (int t = 0; t < T_; t++) {
        size_t base = ((size_t)t * H_ + h) * 64;
        float k0v = k[base + lane], k1v = k[base + lane + 32];
        float q0v = q[base + lane], q1v = q[base + lane + 32];
        float e0  = __expf(g[base + lane]);
        float e1  = __expf(g[base + lane + 32]);
        float vv  = v[base + dv0 + (lane & 7)];
        float bt  = beta[t * H_ + h];
        float oval = 0.f;
#pragma unroll
        for (int c = 0; c < 8; c++) {
            S[c][0] *= e0; S[c][1] *= e1;
            float part = k0v * S[c][0] + k1v * S[c][1];
#pragma unroll
            for (int m = 16; m >= 1; m >>= 1) part += __shfl_xor(part, m, 32);
            float vc = __shfl(vv, c, 32);
            float delta = (vc - part) * bt;
            S[c][0] += k0v * delta; S[c][1] += k1v * delta;
            float op = q0v * S[c][0] + q1v * S[c][1];
#pragma unroll
            for (int m = 16; m >= 1; m >>= 1) op += __shfl_xor(op, m, 32);
            if (lane == c) oval = op;
        }
        if (lane < 8) o[base + dv0 + lane] = oval;
    }
}

// ---------------------------------------------------------------- launcher
extern "C" void kernel_launch(void* const* d_in, const int* in_sizes, int n_in,
                              void* d_out, int out_size, void* d_ws, size_t ws_size,
                              hipStream_t stream) {
    const float* x       = (const float*)d_in[0];
    const float* Wq      = (const float*)d_in[1];
    const float* Wk      = (const float*)d_in[2];
    const float* Wv      = (const float*)d_in[3];
    const float* Wb      = (const float*)d_in[4];
    const float* Wfa     = (const float*)d_in[5];
    const float* Wfb     = (const float*)d_in[6];
    const float* dt_bias = (const float*)d_in[7];
    const float* A_log   = (const float*)d_in[8];
    const float* Wga     = (const float*)d_in[9];
    const float* Wgb     = (const float*)d_in[10];
    const float* conv_q  = (const float*)d_in[11];
    const float* conv_k  = (const float*)d_in[12];
    const float* conv_v  = (const float*)d_in[13];
    const float* o_w     = (const float*)d_in[14];
    const float* Wo      = (const float*)d_in[15];
    float* out = (float*)d_out;

    char* ws = (char*)d_ws;
    size_t off = 0;
    auto alloc = [&](size_t bytes) -> void* {
        void* p = ws + off;
        off += (bytes + 255) & ~(size_t)255;
        return p;
    };
    __hip_bfloat16* xb  = (__hip_bfloat16*)alloc((size_t)T_ * HID_ * 2);
    __hip_bfloat16* wqb = (__hip_bfloat16*)alloc((size_t)HID_ * P_ * 2);
    __hip_bfloat16* wkb = (__hip_bfloat16*)alloc((size_t)HID_ * P_ * 2);
    __hip_bfloat16* wvb = (__hip_bfloat16*)alloc((size_t)HID_ * P_ * 2);
    __hip_bfloat16* wob = (__hip_bfloat16*)alloc((size_t)P_ * HID_ * 2);
    __hip_bfloat16* ob  = (__hip_bfloat16*)alloc((size_t)T_ * P_ * 2);
    float* qp   = (float*)alloc((size_t)T_ * P_ * 4);
    float* kp   = (float*)alloc((size_t)T_ * P_ * 4);
    float* vp   = (float*)alloc((size_t)T_ * P_ * 4);
    float* qc   = (float*)alloc((size_t)T_ * P_ * 4);
    float* kc   = (float*)alloc((size_t)T_ * P_ * 4);
    float* vc   = (float*)alloc((size_t)T_ * P_ * 4);
    float* gl   = (float*)alloc((size_t)T_ * P_ * 4);
    float* g2l  = (float*)alloc((size_t)T_ * P_ * 4);
    float* beta = (float*)alloc((size_t)T_ * H_ * 4);
    float* xfa  = (float*)alloc((size_t)T_ * 64 * 4);
    float* xga  = (float*)alloc((size_t)T_ * 64 * 4);
    float* osc  = (float*)alloc((size_t)T_ * P_ * 4);

    const int TP = T_ * P_;
    // f32 -> bf16 conversions
    k_cvt_bf16<<<(T_ * HID_ + 255) / 256, 256, 0, stream>>>(x, xb, T_ * HID_);
    k_cvt_bf16<<<(HID_ * P_ + 255) / 256, 256, 0, stream>>>(Wq, wqb, HID_ * P_);
    k_cvt_bf16<<<(HID_ * P_ + 255) / 256, 256, 0, stream>>>(Wk, wkb, HID_ * P_);
    k_cvt_bf16<<<(HID_ * P_ + 255) / 256, 256, 0, stream>>>(Wv, wvb, HID_ * P_);
    k_cvt_bf16<<<(P_ * HID_ + 255) / 256, 256, 0, stream>>>(Wo, wob, P_ * HID_);

    // big bf16 WMMA GEMMs: q/k/v projections
    dim3 gQKV(P_ / 128, T_ / 128);
    k_gemm_bf16<<<gQKV, 256, 0, stream>>>(xb, wqb, qp, T_, P_, HID_);
    k_gemm_bf16<<<gQKV, 256, 0, stream>>>(xb, wkb, kp, T_, P_, HID_);
    k_gemm_bf16<<<gQKV, 256, 0, stream>>>(xb, wvb, vp, T_, P_, HID_);

    // causal conv + silu
    k_conv_silu<<<TP / 256, 256, 0, stream>>>(qp, conv_q, qc);
    k_conv_silu<<<TP / 256, 256, 0, stream>>>(kp, conv_k, kc);
    k_conv_silu<<<TP / 256, 256, 0, stream>>>(vp, conv_v, vc);

    // beta / low-rank gates
    k_skinny<<<T_ / 64, 256, 0, stream>>>(x, Wb, beta, HID_, H_, 1);
    k_skinny<<<T_ / 64, 256, 0, stream>>>(x, Wfa, xfa, HID_, 64, 0);
    k_skinny<<<T_ / 64, 256, 0, stream>>>(x, Wga, xga, HID_, 64, 0);
    dim3 gExp(P_ / 128, T_ / 16);
    k_expand<<<gExp, 256, 0, stream>>>(xfa, Wfb, gl);
    k_expand<<<gExp, 256, 0, stream>>>(xga, Wgb, g2l);
    k_gate<<<TP / 256, 256, 0, stream>>>(gl, dt_bias, A_log);

    // q/k L2 norm (q scaled by D^-0.5 = 0.125)
    k_l2norm<<<(T_ * H_) / 8, 256, 0, stream>>>(qc, 0.125f);
    k_l2norm<<<(T_ * H_) / 8, 256, 0, stream>>>(kc, 1.0f);

    // KDA delta-rule scan: 32 heads x 8 dv-groups = 256 waves
    k_scan<<<H_ * 8, 32, 0, stream>>>(qc, kc, vc, gl, beta, osc);

    // gated RMSNorm -> bf16, then output projection
    k_outnorm<<<(T_ * H_) / 8, 256, 0, stream>>>(osc, g2l, o_w, ob);
    dim3 gO(HID_ / 128, T_ / 128);
    k_gemm_bf16<<<gO, 256, 0, stream>>>(ob, wob, out, T_, HID_, P_);

    (void)in_sizes; (void)n_in; (void)out_size; (void)ws_size;
}
